// VarianceAdaptor_1932735283444
// MI455X (gfx1250) — matrix operand order repaired and checked by reading, not compiled, not run on hardware
//
#include <hip/hip_runtime.h>

typedef unsigned short us16;
typedef unsigned int   u32;
typedef __attribute__((ext_vector_type(4)))  unsigned int v4u;
typedef __attribute__((ext_vector_type(8)))  float        v8f;
typedef __attribute__((ext_vector_type(16))) __bf16       v16bf;

#define BB   64
#define SS   512
#define DD   256
#define ML   2048
#define NEMO 7

__device__ __forceinline__ us16 f2bf(float x) {
  u32 u = __float_as_uint(x);
  u32 r = u + 0x7FFFu + ((u >> 16) & 1u);   // round-to-nearest-even
  return (us16)(r >> 16);
}

// ---------------- conversion kernels ----------------
__global__ __launch_bounds__(256) void cvt_bf16_kernel(
    const float* __restrict__ in, us16* __restrict__ out, int n) {
  int i = blockIdx.x * 256 + threadIdx.x;
  if (i < n) out[i] = f2bf(in[i]);
}

// conv weight (3,256,256)[tap][k][n] f32 -> bf16 transposed [tap][n][k]
__global__ __launch_bounds__(256) void cvt_wT_kernel(
    const float* __restrict__ w, us16* __restrict__ wT) {
  int i = blockIdx.x * 256 + threadIdx.x;      // 0 .. 3*256*256-1
  int tap = i >> 16;
  int rem = i & 65535;
  int n = rem >> 8;
  int k = rem & 255;
  wT[i] = f2bf(w[tap * 65536 + k * 256 + n]);
}

// ---------------- fused conv(K=3) + bias + relu + layernorm ----------------
// Block: 256 threads = 8 waves; tile = 64 rows x 256 ch.
// Wave w owns 32 channels (2 N-frags) x all 64 rows (4 M-frags) -> each B
// fragment feeds 4 WMMAs (4x register reuse of weights). 24 k-steps
// (3 taps x 8 chunks of 32) with ping-pong double-buffered fragment prefetch
// (no rotation moves: each buffer set is loaded straight into its consume regs).
#define TMR  64
#define ASTR 264   // LDS A row stride in ushorts (528B: 16B-group stride odd)
#define CSTR 260   // LDS C row stride in floats
#define SMEM_BYTES 66560   // max(66 rows * 528B = 34848, 64 * 260 * 4 = 66560)

__global__ __launch_bounds__(256)
void conv_block_kernel(const us16* __restrict__ xb,   // (B,T,256) bf16
                       const us16* __restrict__ wT,   // (3,256,256) bf16 [tap][n][k]
                       const float* __restrict__ bias,
                       const float* __restrict__ gmm,
                       const float* __restrict__ bta,
                       us16* __restrict__ out_bf,     // nullable
                       float* __restrict__ out_f,     // nullable
                       int T) {
  __shared__ __align__(16) unsigned char smem[SMEM_BYTES];
  us16*  ldsA = (us16*)smem;
  float* ldsC = (float*)smem;

  const int tiles = T >> 6;
  const int b   = blockIdx.x / tiles;
  const int t0  = (blockIdx.x - b * tiles) << 6;
  const int tid = threadIdx.x;
  const int lane = tid & 31;
  const int wave = tid >> 5;

  // stage A: rows t0-1 .. t0+64 (66 rows x 256 bf16), 16B vector copies
  {
    const v4u* xs = (const v4u*)xb;         // 32 x 16B per row
    for (int f = tid; f < 66 * 32; f += 256) {
      int r = f >> 5, c4 = f & 31;
      int t = t0 - 1 + r;
      v4u v = {0u, 0u, 0u, 0u};
      if (t >= 0 && t < T) v = xs[(size_t)(b * T + t) * 32 + c4];
      *(v4u*)(ldsA + r * ASTR + c4 * 8) = v;
    }
  }
  __syncthreads();

  const int m15   = lane & 15;
  const int kHalf = (lane & 16) >> 1;   // A frag K offset (elements)
  const int bHalf = (lane & 16) >> 1;   // B frag K offset (elements)
  const int ncol0 = (wave << 5) + m15;  // this wave's first output channel

  v8f acc[8];                            // [mf*2 + nl]
  for (int i = 0; i < 8; ++i)
    for (int e = 0; e < 8; ++e) acc[i][e] = 0.f;

  // fragment loaders: ks = tap*8 + kc
  auto loadKS = [&](int ks, v16bf aF[4], v16bf bF[2]) {
    const int tap = ks >> 3;
    const int kb  = (ks & 7) << 5;
#pragma unroll
    for (int mf = 0; mf < 4; ++mf) {
      const us16* p = ldsA + ((mf << 4) + m15 + tap) * ASTR + kb + kHalf;
      union { v4u q[2]; v16bf v; } u;
      u.q[0] = *(const v4u*)p;
      u.q[1] = *(const v4u*)(p + 16);
      aF[mf] = u.v;
    }
#pragma unroll
    for (int nl = 0; nl < 2; ++nl) {
      const us16* p = wT + (tap << 16) + ((ncol0 + (nl << 4)) << 8) + kb + bHalf;
      union { v4u q[2]; v16bf v; } u;
      u.q[0] = *(const v4u*)p;
      u.q[1] = *(const v4u*)(p + 16);
      bF[nl] = u.v;
    }
  };
  auto doWMMA = [&](v16bf aF[4], v16bf bF[2]) {
#pragma unroll
    for (int nl = 0; nl < 2; ++nl)
#pragma unroll
      for (int mf = 0; mf < 4; ++mf)
        acc[mf * 2 + nl] = __builtin_amdgcn_wmma_f32_16x16x32_bf16(
            false, aF[mf], false, bF[nl], (short)0, acc[mf * 2 + nl],
            false, false);
  };

  // ping-pong double buffer over 24 k-steps
  v16bf a0[4], b0[2], a1[4], b1[2];
  loadKS(0, a0, b0);
  for (int ks = 0; ks < 22; ks += 2) {
    loadKS(ks + 1, a1, b1);   // prefetch odd step
    doWMMA(a0, b0);           // consume even step
    loadKS(ks + 2, a0, b0);   // prefetch next even step
    doWMMA(a1, b1);           // consume odd step
  }
  loadKS(23, a1, b1);
  doWMMA(a0, b0);             // k-step 22
  doWMMA(a1, b1);             // k-step 23
  __syncthreads();   // done reading ldsA; reuse buffer as ldsC

  // bias + relu -> LDS C tile (64 x 256)
  const int rOff = (lane & 16) ? 8 : 0;
  for (int nl = 0; nl < 2; ++nl) {
    const int n = (wave << 5) + (nl << 4) + m15;
    const float bv = bias[n];
#pragma unroll
    for (int mf = 0; mf < 4; ++mf)
#pragma unroll
      for (int r = 0; r < 8; ++r) {
        float v = acc[mf * 2 + nl][r] + bv;
        ldsC[((mf << 4) + rOff + r) * CSTR + n] = v > 0.f ? v : 0.f;
      }
  }
  __syncthreads();

  // layernorm: 8 rows per wave, lane-parallel across 256 channels
  for (int rr = 0; rr < 8; ++rr) {
    const int row = (wave << 3) + rr;
    float vals[8], s = 0.f, sq = 0.f;
#pragma unroll
    for (int j = 0; j < 8; ++j) {
      float v = ldsC[row * CSTR + lane + (j << 5)];
      vals[j] = v; s += v; sq += v * v;
    }
#pragma unroll
    for (int off = 16; off > 0; off >>= 1) {
      s  += __shfl_xor(s, off, 32);
      sq += __shfl_xor(sq, off, 32);
    }
    const float mean = s * (1.f / 256.f);
    const float var  = sq * (1.f / 256.f) - mean * mean;
    const float rstd = rsqrtf(var + 1e-5f);
    const size_t base = ((size_t)(b * T + t0 + row)) << 8;
#pragma unroll
    for (int j = 0; j < 8; ++j) {
      const int c = lane + (j << 5);
      const float y = (vals[j] - mean) * rstd * gmm[c] + bta[c];
      if (out_f)  out_f[base + c]  = y;
      if (out_bf) out_bf[base + c] = f2bf(y);
    }
  }
}

// ---------------- linear heads ----------------
__global__ __launch_bounds__(256)
void proj_dur_kernel(const float* __restrict__ h, const float* __restrict__ lw,
                     const float* __restrict__ lb, const int* __restrict__ mask,
                     float* __restrict__ log_dur) {
  const int row  = blockIdx.x * 8 + (threadIdx.x >> 5);
  const int lane = threadIdx.x & 31;
  const float* hp = h + ((size_t)row << 8);
  float s = 0.f;
#pragma unroll
  for (int j = 0; j < 8; ++j) { int c = lane + (j << 5); s += hp[c] * lw[c]; }
  for (int off = 16; off > 0; off >>= 1) s += __shfl_xor(s, off, 32);
  if (lane == 0) {
    float v = (s + lb[0]) * (float)mask[row];
    log_dur[row] = fminf(fmaxf(v, -5.f), 5.f);
  }
}

__global__ __launch_bounds__(256)
void proj_masked_kernel(const float* __restrict__ h, const float* __restrict__ lw,
                        const float* __restrict__ lb, const int* __restrict__ lenI,
                        float* __restrict__ pred) {
  const int row  = blockIdx.x * 8 + (threadIdx.x >> 5);
  const int lane = threadIdx.x & 31;
  const float* hp = h + ((size_t)row << 8);
  float s = 0.f;
#pragma unroll
  for (int j = 0; j < 8; ++j) { int c = lane + (j << 5); s += hp[c] * lw[c]; }
  for (int off = 16; off > 0; off >>= 1) s += __shfl_xor(s, off, 32);
  if (lane == 0) {
    const int bb = row >> 11, t = row & 2047;     // T == 2048 here
    const float m = (t < lenI[bb]) ? 1.f : 0.f;
    pred[row] = (s + lb[0]) * m;
  }
}

__global__ __launch_bounds__(256)
void proj_emo_kernel(const float* __restrict__ h, const float* __restrict__ lw,
                     const float* __restrict__ lb, float* __restrict__ logits) {
  const int row  = blockIdx.x * 8 + (threadIdx.x >> 5);
  const int lane = threadIdx.x & 31;
  const float* hp = h + ((size_t)row << 8);
  float pe[NEMO];
  for (int e = 0; e < NEMO; ++e) pe[e] = 0.f;
#pragma unroll
  for (int j = 0; j < 8; ++j) {
    const int c = lane + (j << 5);
    const float v = hp[c];
    for (int e = 0; e < NEMO; ++e) pe[e] += v * lw[c * NEMO + e];
  }
  for (int off = 16; off > 0; off >>= 1)
    for (int e = 0; e < NEMO; ++e) pe[e] += __shfl_xor(pe[e], off, 32);
  if (lane == 0)
    for (int e = 0; e < NEMO; ++e)
      logits[(size_t)row * NEMO + e] = pe[e] + lb[e];
}

// ---------------- emotion: masked mean + per-position argmax ----------------
__global__ __launch_bounds__(256)
void emo_reduce_kernel(const float* __restrict__ logits, const int* __restrict__ mask,
                       int* __restrict__ eidx, float* __restrict__ emo_pred) {
  __shared__ float part[256 * 8];
  const int b = blockIdx.x, tid = threadIdx.x;
  float loc[8];
  for (int e = 0; e < 8; ++e) loc[e] = 0.f;
  for (int t = tid; t < SS; t += 256) {
    const float* lg = logits + ((size_t)(b * SS + t)) * NEMO;
    const float mk = (float)mask[b * SS + t];
    int best = 0; float bv = lg[0];
    for (int e = 1; e < NEMO; ++e) { float v = lg[e]; if (v > bv) { bv = v; best = e; } }
    eidx[b * SS + t] = best;
    for (int e = 0; e < NEMO; ++e) loc[e] += lg[e] * mk;
    loc[7] += mk;
  }
  for (int e = 0; e < 8; ++e) part[tid * 8 + e] = loc[e];
  __syncthreads();
  for (int st = 128; st > 0; st >>= 1) {
    if (tid < st)
      for (int e = 0; e < 8; ++e) part[tid * 8 + e] += part[(tid + st) * 8 + e];
    __syncthreads();
  }
  if (tid < NEMO) emo_pred[b * NEMO + tid] = part[tid] / fmaxf(part[7], 1.f);
}

// ---------------- duration rounding + inclusive scan ----------------
__global__ __launch_bounds__(512)
void cumsum_kernel(const float* __restrict__ log_dur, const int* __restrict__ maxlen,
                   int* __restrict__ ends, int* __restrict__ lenI,
                   float* __restrict__ lenF) {
  __shared__ int sc[512];
  const int b = blockIdx.x, t = threadIdx.x;
  const float ld = log_dur[b * SS + t];
  const float dp = fmaxf(__expf(ld) - 1.f, 0.f);
  const float dr = fminf(fmaxf(rintf(dp), 0.f), 300.f);
  sc[t] = (int)dr;
  __syncthreads();
  for (int off = 1; off < 512; off <<= 1) {
    const int add = (t >= off) ? sc[t - off] : 0;
    __syncthreads();
    sc[t] += add;
    __syncthreads();
  }
  ends[b * SS + t] = sc[t];
  if (t == 511) {
    const int L = min(sc[511], maxlen[0]);
    lenI[b] = L;
    lenF[b] = (float)L;
  }
}

// ---------------- length regulate (gather) + emotion embedding add ----------------
__global__ __launch_bounds__(256)
void regulate_kernel(const float* __restrict__ x, const float* __restrict__ emo_emb,
                     const int* __restrict__ eidx, const int* __restrict__ ends,
                     const int* __restrict__ lenI, float* __restrict__ outF,
                     us16* __restrict__ outB) {
  const int b = blockIdx.x >> 11;
  const int m = blockIdx.x & 2047;
  const int* e = ends + b * SS;
  int lo = 0, hi = SS;                       // searchsorted(side='right')
  while (lo < hi) { const int mid = (lo + hi) >> 1; if (e[mid] <= m) lo = mid + 1; else hi = mid; }
  const int idx = min(lo, SS - 1);
  const bool valid = m < lenI[b];
  const int c = threadIdx.x;
  float v = 0.f;
  if (valid) {
    const int ei = eidx[b * SS + idx];
    v = x[((size_t)(b * SS + idx) << 8) + c] + emo_emb[ei * DD + c];
  }
  const size_t pos = ((size_t)(b * ML + m) << 8) + c;
  outF[pos] = v;
  outB[pos] = f2bf(v);
}

// ---------------- bucketize + embedding add (pitch mode=0, energy mode=1) ----------------
__global__ __launch_bounds__(256)
void add_emb_kernel(float* __restrict__ outF, us16* __restrict__ outB,
                    const float* __restrict__ pred, const float* __restrict__ emb,
                    int mode) {
  const int b = blockIdx.x >> 11;
  const int m = blockIdx.x & 2047;
  const float p = pred[b * ML + m];
  float u;
  if (mode == 0) { const float v = fminf(fmaxf(p, -3.f), 3.f); u = (v + 3.f) * (254.f / 6.f); }
  else           { const float v = fminf(fmaxf(p, 0.f), 1.f);  u = v * 254.f; }
  int idx = (int)ceilf(u);
  idx = min(max(idx, 0), 255);
  const int c = threadIdx.x;
  const size_t pos = ((size_t)(b * ML + m) << 8) + c;
  const float nv = outF[pos] + emb[idx * DD + c];
  outF[pos] = nv;
  if (outB) outB[pos] = f2bf(nv);
}

// ---------------- host launcher ----------------
extern "C" void kernel_launch(void* const* d_in, const int* in_sizes, int n_in,
                              void* d_out, int out_size, void* d_ws, size_t ws_size,
                              hipStream_t stream) {
  (void)in_sizes; (void)n_in; (void)out_size; (void)ws_size;
  auto F = [&](int i) { return (const float*)d_in[i]; };
  const float* x      = F(0);
  const int*   mask   = (const int*)d_in[1];
  const int*   maxlen = (const int*)d_in[2];
  const int DUR = 3, PIT = 13, ENE = 23, EMO = 33;   // w1 b1 g1 be1 w2 b2 g2 be2 lw lb
  const float* pitch_emb  = F(43);
  const float* energy_emb = F(44);
  const float* emo_emb    = F(45);

  // workspace carve-up (256B aligned)
  unsigned char* w = (unsigned char*)d_ws;
  size_t o = 0;
  auto take = [&](size_t bytes) -> unsigned char* {
    unsigned char* p = w + o;
    o = (o + bytes + 255) & ~(size_t)255;
    return p;
  };
  us16*  XB   = (us16*)take((size_t)BB * SS * DD * 2);
  us16*  WT[8];
  for (int i = 0; i < 8; ++i) WT[i] = (us16*)take((size_t)3 * DD * DD * 2);
  us16*  H1B  = (us16*)take((size_t)BB * ML * DD * 2);
  float* H2F  = (float*)take((size_t)BB * ML * DD * 4);
  us16*  OUTB = (us16*)take((size_t)BB * ML * DD * 2);
  float* ELOG = (float*)take((size_t)BB * SS * NEMO * 4);
  int*   EIDX = (int*)take((size_t)BB * SS * 4);
  int*   ENDS = (int*)take((size_t)BB * SS * 4);
  int*   LENI = (int*)take((size_t)BB * 4);

  // output layout (flat concat, return order)
  float* OUT    = (float*)d_out;                 // (B, 2048, 256)
  float* LOGDUR = OUT + (size_t)BB * ML * DD;    // (B, 512)
  float* PP     = LOGDUR + (size_t)BB * SS;      // (B, 2048)
  float* EP     = PP + (size_t)BB * ML;          // (B, 2048)
  float* LENF   = EP + (size_t)BB * ML;          // (B,)
  float* EMOP   = LENF + BB;                     // (B, 7)

  // conversions
  { int n = BB * SS * DD; cvt_bf16_kernel<<<(n + 255) / 256, 256, 0, stream>>>(x, XB, n); }
  {
    const int widx[8] = {DUR + 0, DUR + 4, EMO + 0, EMO + 4,
                         PIT + 0, PIT + 4, ENE + 0, ENE + 4};
    for (int i = 0; i < 8; ++i)
      cvt_wT_kernel<<<(3 * DD * DD) / 256, 256, 0, stream>>>(F(widx[i]), WT[i]);
  }
  const int blksS = BB * (SS / TMR), blksM = BB * (ML / TMR);

  // duration predictor (on x, T=512)
  conv_block_kernel<<<blksS, 256, 0, stream>>>(XB,  WT[0], F(DUR + 1), F(DUR + 2), F(DUR + 3), H1B, nullptr, SS);
  conv_block_kernel<<<blksS, 256, 0, stream>>>(H1B, WT[1], F(DUR + 5), F(DUR + 6), F(DUR + 7), nullptr, H2F, SS);
  proj_dur_kernel<<<BB * SS / 8, 256, 0, stream>>>(H2F, F(DUR + 8), F(DUR + 9), mask, LOGDUR);

  // emotion predictor (on x, T=512)
  conv_block_kernel<<<blksS, 256, 0, stream>>>(XB,  WT[2], F(EMO + 1), F(EMO + 2), F(EMO + 3), H1B, nullptr, SS);
  conv_block_kernel<<<blksS, 256, 0, stream>>>(H1B, WT[3], F(EMO + 5), F(EMO + 6), F(EMO + 7), nullptr, H2F, SS);
  proj_emo_kernel<<<BB * SS / 8, 256, 0, stream>>>(H2F, F(EMO + 8), F(EMO + 9), ELOG);
  emo_reduce_kernel<<<BB, 256, 0, stream>>>(ELOG, mask, EIDX, EMOP);

  // durations -> ends/lengths, then length-regulate gather (+ emo embedding)
  cumsum_kernel<<<BB, 512, 0, stream>>>(LOGDUR, maxlen, ENDS, LENI, LENF);
  regulate_kernel<<<BB * ML, 256, 0, stream>>>(x, emo_emb, EIDX, ENDS, LENI, OUT, OUTB);

  // pitch predictor (on out, T=2048)
  conv_block_kernel<<<blksM, 256, 0, stream>>>(OUTB, WT[4], F(PIT + 1), F(PIT + 2), F(PIT + 3), H1B, nullptr, ML);
  conv_block_kernel<<<blksM, 256, 0, stream>>>(H1B,  WT[5], F(PIT + 5), F(PIT + 6), F(PIT + 7), nullptr, H2F, ML);
  proj_masked_kernel<<<BB * ML / 8, 256, 0, stream>>>(H2F, F(PIT + 8), F(PIT + 9), LENI, PP);
  add_emb_kernel<<<BB * ML, 256, 0, stream>>>(OUT, OUTB, PP, pitch_emb, 0);

  // energy predictor (on out, T=2048)
  conv_block_kernel<<<blksM, 256, 0, stream>>>(OUTB, WT[6], F(ENE + 1), F(ENE + 2), F(ENE + 3), H1B, nullptr, ML);
  conv_block_kernel<<<blksM, 256, 0, stream>>>(H1B,  WT[7], F(ENE + 5), F(ENE + 6), F(ENE + 7), nullptr, H2F, ML);
  proj_masked_kernel<<<BB * ML / 8, 256, 0, stream>>>(H2F, F(ENE + 8), F(ENE + 9), LENI, EP);
  add_emb_kernel<<<BB * ML, 256, 0, stream>>>(OUT, nullptr, EP, energy_emb, 1);
}
